// CostComputer_6219112645429
// MI455X (gfx1250) — compile-verified, matrix operand order checked by reading, not run on hardware
//
#include <hip/hip_runtime.h>
#include <math.h>

typedef __attribute__((ext_vector_type(16))) _Float16 v16h;
typedef __attribute__((ext_vector_type(8)))  float    v8f;

#define B_TOT   16384
#define NDIM    256
#define NQ      16
#define NS      16
#define SOCM    128
#define NCOLS   6752        // 512 + 16*256 + 16*128 + 16 + 16 + 64
#define NTILES  422         // NCOLS / 16
#define KFRAGS  8           // 256 / 32
// tile-index segment boundaries
#define T_SEG1  32          // A1 tiles: [0,32)
#define T_SEG2  288         // P  tiles: [32,288)
#define T_SEG3  416         // M  tiles: [288,416)
#define T_SEG4  417         // c  tile : 416
#define T_SEG5  418         // q  tile : 417 ; A2 tiles: [418,422)

#define WPK_HALVES ((size_t)NTILES * KFRAGS * 32 * 16)

#define WAVES   2
#define YSTRIDE 272         // 272 % 64 == 16 -> upper half-wave hits disjoint banks

// ---------------- pack W (concat of A1,P,M,c,q,A2) into B-fragment-major f16 ----------------
__global__ void pack_w(const float* __restrict__ A1, const float* __restrict__ P,
                       const float* __restrict__ M,  const float* __restrict__ c,
                       const float* __restrict__ q,  const float* __restrict__ A2,
                       _Float16* __restrict__ wpk) {
  int tid = blockIdx.x * blockDim.x + threadIdx.x;
  if (tid >= NTILES * KFRAGS * 32) return;
  int lane = tid & 31;
  int f    = (tid >> 5) & 7;
  int ct   = tid >> 8;
  int col  = ct * 16 + (lane & 15);
  int kBase = f * 32 + ((lane & 16) ? 8 : 0);
  _Float16* dst = wpk + (size_t)tid * 16;
#pragma unroll
  for (int h = 0; h < 16; ++h) {
    int k = kBase + (h < 8 ? h : h + 8);   // 16-bit fragment K pattern
    float v;
    if      (col < 512)  { v = A1[col * NDIM + k]; }
    else if (col < 4608) { int cc = col - 512;  v = P[((cc >> 8) * NDIM + (cc & 255)) * NDIM + k]; }
    else if (col < 6656) { int cc = col - 4608; v = M[((cc >> 7) * SOCM + (cc & 127)) * NDIM + k]; }
    else if (col < 6672) { v = c[(col - 6656) * NDIM + k]; }
    else if (col < 6688) { v = q[(col - 6672) * NDIM + k]; }
    else                 { v = A2[(col - 6688) * NDIM + k]; }
    dst[h] = (_Float16)v;
  }
}

// additive bias per output column: lin -> -b1, SOC rows -> +s, eq -> -b2, else 0
__global__ void pack_bias(const float* __restrict__ b1, const float* __restrict__ s,
                          const float* __restrict__ b2, float* __restrict__ bias) {
  int col = blockIdx.x * blockDim.x + threadIdx.x;
  if (col >= NCOLS) return;
  float v = 0.f;
  if      (col < 512)  v = -b1[col];
  else if (col < 4608) v = 0.f;
  else if (col < 6656) { int cc = col - 4608; v = s[(cc >> 7) * SOCM + (cc & 127)]; }
  else if (col < 6688) v = 0.f;
  else                 v = -b2[col - 6688];
  bias[col] = v;
}

__global__ void zero_out(float* o) { if (threadIdx.x == 0 && blockIdx.x == 0) o[0] = 0.f; }

__device__ __forceinline__ float red16(float v) {   // sum across 16-lane column group
  v += __shfl_xor(v, 1, 16);
  v += __shfl_xor(v, 2, 16);
  v += __shfl_xor(v, 4, 16);
  v += __shfl_xor(v, 8, 16);
  return v;
}

// ---------------- fused GEMM + penalty epilogue ----------------
__global__ __launch_bounds__(WAVES * 32) void fused_gemm(
    const float* __restrict__ y, const _Float16* __restrict__ wpk,
    const float* __restrict__ bias, const float* __restrict__ rr,
    const float* __restrict__ dd, float* __restrict__ out) {

  __shared__ float yT[WAVES][16][YSTRIDE];
  __shared__ float accQ[WAVES][16][NQ];
  __shared__ float accS[WAVES][16][NS];
  __shared__ float cdot[WAVES][16][NS];
  __shared__ float qdot[WAVES][16][NQ];

  const int lane = threadIdx.x & 31;
  const int wid  = threadIdx.x >> 5;
  const int rowBase = (blockIdx.x * WAVES + wid) * 16;

  // stage this wave's 16 y rows (f32) into LDS
  for (int i = lane; i < 16 * 64; i += 32) {
    int r = i >> 6, cq = i & 63;
    float4 v = ((const float4*)(y + (size_t)(rowBase + r) * NDIM))[cq];
    yT[wid][r][cq * 4 + 0] = v.x; yT[wid][r][cq * 4 + 1] = v.y;
    yT[wid][r][cq * 4 + 2] = v.z; yT[wid][r][cq * 4 + 3] = v.w;
  }
  __syncthreads();

  // resident A fragments (16x32 f16 each), per ISA 16-bit A layout
  const int rowA = lane & 15;
  const int kOff = (lane & 16) ? 8 : 0;
  v16h a[KFRAGS];
#pragma unroll
  for (int f = 0; f < KFRAGS; ++f)
#pragma unroll
    for (int h = 0; h < 16; ++h)
      a[f][h] = (_Float16)yT[wid][rowA][f * 32 + kOff + (h < 8 ? h : h + 8)];

  const int rLoc = (lane >> 4) * 8;   // row offset of this half-wave's D elements
  float lsum = 0.f;

  auto mm = [&](int ct) -> v8f {
    const v16h* bp = (const v16h*)wpk + (size_t)ct * KFRAGS * 32 + lane;
    v8f acc = {};
#pragma unroll
    for (int f = 0; f < KFRAGS; ++f) {
      v16h bfr = bp[f * 32];
      acc = __builtin_amdgcn_wmma_f32_16x16x32_f16(false, a[f], false, bfr,
                                                   (short)0, acc, false, false);
    }
    return acc;
  };

  // seg0: linear inequalities, sum relu(D - b1)^2
  for (int ct = 0; ct < T_SEG1; ++ct) {
    v8f d = mm(ct);
    float bv = bias[ct * 16 + (lane & 15)];
#pragma unroll
    for (int r = 0; r < 8; ++r) { float v = fmaxf(d[r] + bv, 0.f); lsum += v * v; }
  }

  // seg1: Py, fused dot with y -> accQ[row][k] = y^T P_k y
  for (int kk = 0; kk < NQ; ++kk) {
    float part[8] = {0.f, 0.f, 0.f, 0.f, 0.f, 0.f, 0.f, 0.f};
    for (int t = 0; t < 16; ++t) {
      int ct = T_SEG1 + kk * 16 + t;
      v8f d = mm(ct);
      int nloc = t * 16 + (lane & 15);
#pragma unroll
      for (int r = 0; r < 8; ++r) part[r] += d[r] * yT[wid][rLoc + r][nloc];
    }
#pragma unroll
    for (int r = 0; r < 8; ++r) {
      float red = red16(part[r]);
      if ((lane & 15) == 0) accQ[wid][rLoc + r][kk] = red;
    }
  }

  // seg2: My+s, fused squared-norm -> accS[row][k] = ||M_k y + s_k||^2
  for (int kk = 0; kk < NS; ++kk) {
    float part[8] = {0.f, 0.f, 0.f, 0.f, 0.f, 0.f, 0.f, 0.f};
    for (int t = 0; t < 8; ++t) {
      int ct = T_SEG2 + kk * 8 + t;
      v8f d = mm(ct);
      float bv = bias[ct * 16 + (lane & 15)];
#pragma unroll
      for (int r = 0; r < 8; ++r) { float v = d[r] + bv; part[r] += v * v; }
    }
#pragma unroll
    for (int r = 0; r < 8; ++r) {
      float red = red16(part[r]);
      if ((lane & 15) == 0) accS[wid][rLoc + r][kk] = red;
    }
  }

  // seg3: c^T y  (16 cols, one per SOC constraint)
  {
    v8f d = mm(T_SEG3);
#pragma unroll
    for (int r = 0; r < 8; ++r) cdot[wid][rLoc + r][lane & 15] = d[r];
  }
  // seg4: q^T y
  {
    v8f d = mm(T_SEG4);
#pragma unroll
    for (int r = 0; r < 8; ++r) qdot[wid][rLoc + r][lane & 15] = d[r];
  }

  // seg5: equalities, sum (D - b2)^2 (no relu)
  for (int ct = T_SEG5; ct < NTILES; ++ct) {
    v8f d = mm(ct);
    float bv = bias[ct * 16 + (lane & 15)];
#pragma unroll
    for (int r = 0; r < 8; ++r) { float v = d[r] + bv; lsum += v * v; }
  }

  __syncthreads();

  // final per-row nonlinearities (lanes 0..15 each own one row)
  if (lane < 16) {
    int row = lane;
#pragma unroll
    for (int k = 0; k < NQ; ++k) {
      float qv = fmaxf(0.5f * accQ[wid][row][k] + qdot[wid][row][k] + rr[k], 0.f);
      lsum += qv * qv;
      float sv = fmaxf(sqrtf(accS[wid][row][k]) - cdot[wid][row][k] - dd[k], 0.f);
      lsum += sv * sv;
    }
  }

  // wave reduce + single atomic
  lsum += __shfl_xor(lsum, 1,  32);
  lsum += __shfl_xor(lsum, 2,  32);
  lsum += __shfl_xor(lsum, 4,  32);
  lsum += __shfl_xor(lsum, 8,  32);
  lsum += __shfl_xor(lsum, 16, 32);
  if (lane == 0) atomicAdd(out, lsum);
}

extern "C" void kernel_launch(void* const* d_in, const int* in_sizes, int n_in,
                              void* d_out, int out_size, void* d_ws, size_t ws_size,
                              hipStream_t stream) {
  const float* y  = (const float*)d_in[0];
  const float* A1 = (const float*)d_in[1];
  const float* b1 = (const float*)d_in[2];
  const float* P  = (const float*)d_in[3];
  const float* q  = (const float*)d_in[4];
  const float* r  = (const float*)d_in[5];
  const float* M  = (const float*)d_in[6];
  const float* s  = (const float*)d_in[7];
  const float* c  = (const float*)d_in[8];
  const float* d  = (const float*)d_in[9];
  const float* A2 = (const float*)d_in[10];
  const float* b2 = (const float*)d_in[11];
  float* out = (float*)d_out;

  _Float16* wpk  = (_Float16*)d_ws;
  float*    bias = (float*)((char*)d_ws + WPK_HALVES * sizeof(_Float16));

  int packThreads = NTILES * KFRAGS * 32;
  pack_w<<<(packThreads + 255) / 256, 256, 0, stream>>>(A1, P, M, c, q, A2, wpk);
  pack_bias<<<(NCOLS + 255) / 256, 256, 0, stream>>>(b1, s, b2, bias);
  zero_out<<<1, 32, 0, stream>>>(out);
  fused_gemm<<<B_TOT / (WAVES * 16), WAVES * 32, 0, stream>>>(y, wpk, bias, r, d, out);
}
